// NucleusMoELayer_27848567947312
// MI455X (gfx1250) — matrix operand order repaired and checked by reading, not compiled, not run on hardware
//
#include <hip/hip_runtime.h>
#include <hip/hip_bf16.h>
#include <math.h>

// ---------------------------------------------------------------------------
// MI455X (gfx1250) MoE layer: router + expert-choice top-k + grouped SwiGLU
// experts + shared expert, all GEMMs on v_wmma_f32_16x16x32_bf16 (wave32).
// Round 3: fix async-to-LDS builtin pointer types (wants AS(1)/AS(3) pointers
// to 16-byte int vectors per hipcc diagnostic).
// ---------------------------------------------------------------------------

typedef __attribute__((ext_vector_type(16))) __bf16 v16bf;
typedef __attribute__((ext_vector_type(8)))  __bf16 v8bf;
typedef __attribute__((ext_vector_type(8)))  float  v8f;
typedef int v4i_ __attribute__((vector_size(4 * sizeof(int))));

#define AS1 __attribute__((address_space(1)))
#define AS3 __attribute__((address_space(3)))

#if __has_builtin(__builtin_amdgcn_global_load_async_to_lds_b128)
#define HAVE_ASYNC_LDS 1
#else
#define HAVE_ASYNC_LDS 0
#endif

__device__ inline void async_wait_all() {
#if __has_builtin(__builtin_amdgcn_s_wait_asynccnt)
  __builtin_amdgcn_s_wait_asynccnt(0);
#else
  asm volatile("s_wait_asynccnt 0x0" ::: "memory");
#endif
}

constexpr int BATCH = 4;
constexpr int SEQ   = 4096;
constexpr int HID   = 1024;
constexpr int MDIM  = 1024;
constexpr int NEXP  = 16;
constexpr int CAPC  = 512;             // ceil(2.0 * 4096 / 16)
constexpr int TOK   = BATCH * SEQ;     // 16384
constexpr int TPE   = BATCH * CAPC;    // 2048 token slots per expert
constexpr float ROUTE_SCALE = 1.0f;

// -------------------------------- helpers ----------------------------------

__device__ inline v8bf cvt8(float4 a, float4 b) {
  v8bf r;
  r[0] = (__bf16)a.x; r[1] = (__bf16)a.y; r[2] = (__bf16)a.z; r[3] = (__bf16)a.w;
  r[4] = (__bf16)b.x; r[5] = (__bf16)b.y; r[6] = (__bf16)b.z; r[7] = (__bf16)b.w;
  return r;
}

__device__ inline v16bf cat16(v8bf lo, v8bf hi) {
  return __builtin_shufflevector(lo, hi, 0, 1, 2, 3, 4, 5, 6, 7,
                                 8, 9, 10, 11, 12, 13, 14, 15);
}

__device__ inline v8f wmma_bf16(v16bf a, v16bf b, v8f c) {
  // (neg_a, A, neg_b, B, c_mod, C, reuse_a, reuse_b)
  return __builtin_amdgcn_wmma_f32_16x16x32_bf16(false, a, false, b, (short)0, c,
                                                 false, false);
}

// Fast activations: raw v_rcp_f32 (no div-correction ladder); bf16-accurate.
__device__ inline float sigmoidf_(float x) {
  return __builtin_amdgcn_rcpf(1.0f + __expf(-x));
}
__device__ inline float siluf_(float x) {
  return x * __builtin_amdgcn_rcpf(1.0f + __expf(-x));
}

// Async copy of 16 bytes global -> LDS (falls back to VGPR round-trip).
__device__ inline void copy16_g2l(const __bf16* __restrict__ g, __bf16* l) {
#if HAVE_ASYNC_LDS
  __builtin_amdgcn_global_load_async_to_lds_b128(
      (AS1 v4i_*)(const void*)g, (AS3 v4i_*)(void*)l, 0, 0);
#else
  *(uint4*)l = *(const uint4*)g;
#endif
}

// ------------------------------ router -------------------------------------
__global__ __launch_bounds__(256)
void router_kernel(const float* __restrict__ hsu, const float* __restrict__ ts,
                   const float* __restrict__ gw, float* __restrict__ aff) {
  const int lane = threadIdx.x & 31;
  const int wave = threadIdx.x >> 5;
  const int group = blockIdx.x * 8 + wave;   // 0..1023
  const int tb = group * 16;                 // global token base
  const int b  = tb / SEQ;
  const int l16 = lane & 15, lh = lane >> 4;
  const int row = tb + l16;

  v8f acc = {};
  for (int k0 = 0; k0 < 2 * HID; k0 += 32) {
    const float* src = (k0 < HID) ? (ts + (size_t)b * HID + k0)
                                  : (hsu + (size_t)row * HID + (k0 - HID));
    const float4* s0 = (const float4*)(src + lh * 16);
    const float4* s1 = (const float4*)(src + lh * 16 + 8);
    v16bf afrag = cat16(cvt8(s0[0], s0[1]), cvt8(s1[0], s1[1]));
    v16bf bfrag;
#pragma unroll
    for (int t = 0; t < 16; ++t)
      bfrag[t] = (__bf16)gw[(size_t)(k0 + lh * 16 + t) * NEXP + l16];
    acc = wmma_bf16(afrag, bfrag, acc);
  }
  const int sbase = tb - b * SEQ;
#pragma unroll
  for (int v = 0; v < 8; ++v) {
    int srow = sbase + lh * 8 + v;
    aff[(size_t)b * NEXP * SEQ + (size_t)l16 * SEQ + srow] = sigmoidf_(acc[v]);
  }
}

// --------------------------- expert-choice top-k ---------------------------
__global__ __launch_bounds__(1024)
void topk_kernel(const float* __restrict__ aff, float* __restrict__ gating,
                 int* __restrict__ gidx) {
  const int b = blockIdx.x / NEXP, e = blockIdx.x % NEXP;
  const float* rowp = aff + ((size_t)b * NEXP + e) * SEQ;
  __shared__ unsigned long long keys[SEQ];   // 32 KB
  for (int i = threadIdx.x; i < SEQ; i += blockDim.x) {
    unsigned int bits = __float_as_uint(rowp[i]);  // sigmoid>0 -> monotone bits
    keys[i] = ((unsigned long long)bits << 32) | (unsigned int)i;
  }
  __syncthreads();
  for (int k = 2; k <= SEQ; k <<= 1) {
    for (int j = k >> 1; j > 0; j >>= 1) {
      for (int i = threadIdx.x; i < SEQ; i += blockDim.x) {
        int ixj = i ^ j;
        if (ixj > i) {
          bool desc = ((i & k) == 0);
          unsigned long long a = keys[i], c = keys[ixj];
          if ((a < c) == desc) { keys[i] = c; keys[ixj] = a; }
        }
      }
      __syncthreads();
    }
  }
  for (int c = threadIdx.x; c < CAPC; c += blockDim.x) {
    unsigned long long kk = keys[c];
    int o = e * (BATCH * CAPC) + b * CAPC + c;
    gating[o] = __uint_as_float((unsigned int)(kk >> 32));
    gidx[o]   = b * SEQ + (int)(kk & 0xFFFFFFFFu);
  }
}

// --------------------- gate normalization over experts ---------------------
__global__ void zero_kernel(float* __restrict__ p, int n) {
  int i = blockIdx.x * blockDim.x + threadIdx.x;
  if (i < n) p[i] = 0.0f;
}
__global__ void scatter_sum_kernel(const float* __restrict__ gating,
                                   const int* __restrict__ gidx,
                                   float* __restrict__ sums) {
  int i = blockIdx.x * blockDim.x + threadIdx.x;
  if (i < NEXP * TPE) atomicAdd(&sums[gidx[i]], gating[i]);
}
__global__ void norm_kernel(float* __restrict__ gating,
                            const int* __restrict__ gidx,
                            const float* __restrict__ sums) {
  int i = blockIdx.x * blockDim.x + threadIdx.x;
  if (i < NEXP * TPE)
    gating[i] = gating[i] / (sums[gidx[i]] + 1e-12f) * ROUTE_SCALE;
}

// ------------------------ GEMM1 + fused SwiGLU ------------------------------
__global__ __launch_bounds__(256)
void gemm1_swiglu(const float* __restrict__ A, const int* __restrict__ gidxv,
                  const float* __restrict__ Bw, __bf16* __restrict__ act,
                  int numRowBlocks, int siluFirst) {
  constexpr int BK = 32;
  const int e  = blockIdx.y;
  const int rb = blockIdx.x % numRowBlocks;
  const int cb = blockIdx.x / numRowBlocks;
  const float* Bptr = Bw + (size_t)e * HID * (2 * MDIM);
  __bf16* actPtr = act + (size_t)e * TPE * MDIM;
  const int* gptr = gidxv ? (gidxv + e * TPE) : nullptr;

  __shared__ __bf16 As[128 * BK];   // [row][k]
  __shared__ __bf16 Bg[64 * BK];    // [col][k] (transposed)
  __shared__ __bf16 Bu[64 * BK];

  const int tid = threadIdx.x;
  const int wave = tid >> 5, lane = tid & 31;
  const int l16 = lane & 15, lh = lane >> 4;
  const int rBase = rb * 128, cBase = cb * 64;

  const int ar = tid >> 1, acseg = (tid & 1) * 16;
  int growv = rBase + ar;
  if (gptr) growv = gptr[growv];
  const float* aRow = A + (size_t)growv * HID;
  const int bcol = tid & 63, bkseg = (tid >> 6) * 8;

  v8f accg[4] = {}; v8f accu[4] = {};

  for (int k0 = 0; k0 < HID; k0 += BK) {
    __builtin_prefetch(aRow + k0 + 2 * BK, 0, 1);  // global_prefetch_b8
    {  // A tile (f32 -> bf16)
      const float4* s = (const float4*)(aRow + k0 + acseg);
      float4 f0 = s[0], f1 = s[1], f2 = s[2], f3 = s[3];
      *(v8bf*)(&As[ar * BK + acseg])     = cvt8(f0, f1);
      *(v8bf*)(&As[ar * BK + acseg + 8]) = cvt8(f2, f3);
    }
    {  // B tiles, transposed into LDS
      v8bf pg, pu;
#pragma unroll
      for (int i = 0; i < 8; ++i) {
        size_t kr = (size_t)(k0 + bkseg + i) * (2 * MDIM);
        pg[i] = (__bf16)Bptr[kr + cBase + bcol];
        pu[i] = (__bf16)Bptr[kr + MDIM + cBase + bcol];
      }
      *(v8bf*)(&Bg[bcol * BK + bkseg]) = pg;
      *(v8bf*)(&Bu[bcol * BK + bkseg]) = pu;
    }
    __syncthreads();
    const int arow = wave * 16 + l16;
    v16bf afrag = cat16(*(const v8bf*)(&As[arow * BK + lh * 16]),
                        *(const v8bf*)(&As[arow * BK + lh * 16 + 8]));
#pragma unroll
    for (int nt = 0; nt < 4; ++nt) {
      v16bf bfg = *(const v16bf*)(&Bg[(nt * 16 + l16) * BK + lh * 16]);
      v16bf bfu = *(const v16bf*)(&Bu[(nt * 16 + l16) * BK + lh * 16]);
      accg[nt] = wmma_bf16(afrag, bfg, accg[nt]);
      accu[nt] = wmma_bf16(afrag, bfu, accu[nt]);
    }
    __syncthreads();
  }
#pragma unroll
  for (int nt = 0; nt < 4; ++nt) {
#pragma unroll
    for (int v = 0; v < 8; ++v) {
      float g = accg[nt][v], u = accu[nt][v];
      float x = siluFirst ? siluf_(g) * u : g * siluf_(u);
      int r = rBase + wave * 16 + lh * 8 + v;
      int c = cBase + nt * 16 + l16;
      actPtr[(size_t)r * MDIM + c] = (__bf16)x;
    }
  }
}

// ------------------------- GEMM2 (dense store) ------------------------------
__global__ __launch_bounds__(256)
void gemm2_store(const __bf16* __restrict__ act, const float* __restrict__ Bw,
                 float* __restrict__ out, int numRowBlocks) {
  constexpr int BK = 32;
  const int rb = blockIdx.x % numRowBlocks;
  const int cb = blockIdx.x / numRowBlocks;
  __shared__ __bf16 As[128 * BK];
  __shared__ __bf16 Bt[128 * BK];   // [col][k]
  const int tid = threadIdx.x, wave = tid >> 5, lane = tid & 31;
  const int l16 = lane & 15, lh = lane >> 4;
  const int rBase = rb * 128, cBase = cb * 128;
  const int ar = tid >> 1, acseg = (tid & 1) * 16;
  const int bcol = tid & 127, bhalf = tid >> 7;

  v8f acc[8] = {};
  for (int k0 = 0; k0 < MDIM; k0 += BK) {
    {  // A tile: bf16 copy, async global->LDS when available
      const __bf16* g = act + (size_t)(rBase + ar) * MDIM + k0 + acseg;
      copy16_g2l(g,     &As[ar * BK + acseg]);
      copy16_g2l(g + 8, &As[ar * BK + acseg + 8]);
    }
    {  // B tile transposed, f32 -> bf16
      v8bf p0, p1;
#pragma unroll
      for (int i = 0; i < 8; ++i) {
        p0[i] = (__bf16)Bw[(size_t)(k0 + bhalf * 16 + i) * HID + cBase + bcol];
        p1[i] = (__bf16)Bw[(size_t)(k0 + bhalf * 16 + 8 + i) * HID + cBase + bcol];
      }
      *(v8bf*)(&Bt[bcol * BK + bhalf * 16])     = p0;
      *(v8bf*)(&Bt[bcol * BK + bhalf * 16 + 8]) = p1;
    }
#if HAVE_ASYNC_LDS
    async_wait_all();
#endif
    __syncthreads();
    const int arow = wave * 16 + l16;
    v16bf afrag = cat16(*(const v8bf*)(&As[arow * BK + lh * 16]),
                        *(const v8bf*)(&As[arow * BK + lh * 16 + 8]));
#pragma unroll
    for (int nt = 0; nt < 8; ++nt) {
      v16bf bfrag = *(const v16bf*)(&Bt[(nt * 16 + l16) * BK + lh * 16]);
      acc[nt] = wmma_bf16(afrag, bfrag, acc[nt]);
    }
    __syncthreads();
  }
#pragma unroll
  for (int nt = 0; nt < 8; ++nt)
#pragma unroll
    for (int v = 0; v < 8; ++v) {
      int r = rBase + wave * 16 + lh * 8 + v;
      int c = cBase + nt * 16 + l16;
      out[(size_t)r * HID + c] = acc[nt][v];
    }
}

// --------------------- GEMM2 (scaled atomic scatter) ------------------------
__global__ __launch_bounds__(256)
void gemm2_scatter(const __bf16* __restrict__ act, const float* __restrict__ Bw,
                   const float* __restrict__ gflat, const int* __restrict__ gidx,
                   float* __restrict__ out, int numRowBlocks) {
  constexpr int BK = 32;
  const int e  = blockIdx.y;
  const int rb = blockIdx.x % numRowBlocks;
  const int cb = blockIdx.x / numRowBlocks;
  const __bf16* aBase = act + (size_t)e * TPE * MDIM;
  const float*  bBase = Bw + (size_t)e * MDIM * HID;
  __shared__ __bf16 As[128 * BK];
  __shared__ __bf16 Bt[128 * BK];
  const int tid = threadIdx.x, wave = tid >> 5, lane = tid & 31;
  const int l16 = lane & 15, lh = lane >> 4;
  const int rBase = rb * 128, cBase = cb * 128;
  const int ar = tid >> 1, acseg = (tid & 1) * 16;
  const int bcol = tid & 127, bhalf = tid >> 7;

  v8f acc[8] = {};
  for (int k0 = 0; k0 < MDIM; k0 += BK) {
    {
      const __bf16* g = aBase + (size_t)(rBase + ar) * MDIM + k0 + acseg;
      copy16_g2l(g,     &As[ar * BK + acseg]);
      copy16_g2l(g + 8, &As[ar * BK + acseg + 8]);
    }
    {
      v8bf p0, p1;
#pragma unroll
      for (int i = 0; i < 8; ++i) {
        p0[i] = (__bf16)bBase[(size_t)(k0 + bhalf * 16 + i) * HID + cBase + bcol];
        p1[i] = (__bf16)bBase[(size_t)(k0 + bhalf * 16 + 8 + i) * HID + cBase + bcol];
      }
      *(v8bf*)(&Bt[bcol * BK + bhalf * 16])     = p0;
      *(v8bf*)(&Bt[bcol * BK + bhalf * 16 + 8]) = p1;
    }
#if HAVE_ASYNC_LDS
    async_wait_all();
#endif
    __syncthreads();
    const int arow = wave * 16 + l16;
    v16bf afrag = cat16(*(const v8bf*)(&As[arow * BK + lh * 16]),
                        *(const v8bf*)(&As[arow * BK + lh * 16 + 8]));
#pragma unroll
    for (int nt = 0; nt < 8; ++nt) {
      v16bf bfrag = *(const v16bf*)(&Bt[(nt * 16 + l16) * BK + lh * 16]);
      acc[nt] = wmma_bf16(afrag, bfrag, acc[nt]);
    }
    __syncthreads();
  }
#pragma unroll
  for (int nt = 0; nt < 8; ++nt)
#pragma unroll
    for (int v = 0; v < 8; ++v) {
      int r = rBase + wave * 16 + lh * 8 + v;     // token slot within expert
      int c = cBase + nt * 16 + l16;
      int slot = e * TPE + r;
      float scaled = acc[nt][v] * gflat[slot];
      atomicAdd(&out[(size_t)gidx[slot] * HID + c], scaled);
    }
}

// ------------------------------- launch -------------------------------------
extern "C" void kernel_launch(void* const* d_in, const int* in_sizes, int n_in,
                              void* d_out, int out_size, void* d_ws, size_t ws_size,
                              hipStream_t stream) {
  (void)in_sizes; (void)n_in; (void)out_size; (void)ws_size;
  const float* hs  = (const float*)d_in[0];   // hidden_states [B,S,H]
  const float* hsu = (const float*)d_in[1];   // hidden_states_unmodulated
  const float* ts  = (const float*)d_in[2];   // timestep [B,H]
  const float* gw  = (const float*)d_in[3];   // gate_w [2H,E]
  const float* gup = (const float*)d_in[4];   // gate_up_proj [E,H,2M]
  const float* dwn = (const float*)d_in[5];   // down_proj [E,M,H]
  const float* sgu = (const float*)d_in[6];   // shared_gate_up_w [H,2M]
  const float* sdw = (const float*)d_in[7];   // shared_down_w [M,H]
  float* out = (float*)d_out;

  char* ws = (char*)d_ws;
  float* aff    = (float*)(ws);                                   // 1 MiB
  float* gating = (float*)(ws + 1048576);                         // 128 KiB
  int*   gidx   = (int*)  (ws + 1048576 + 131072);                // 128 KiB
  float* sums   = (float*)(ws + 1048576 + 262144);                // 64 KiB
  __bf16* act   = (__bf16*)(ws + 1048576 + 262144 + 65536);       // 64 MiB (reused)

  router_kernel<<<dim3(TOK / 16 / 8), dim3(256), 0, stream>>>(hsu, ts, gw, aff);
  topk_kernel<<<dim3(BATCH * NEXP), dim3(1024), 0, stream>>>(aff, gating, gidx);
  zero_kernel<<<dim3(TOK / 256), dim3(256), 0, stream>>>(sums, TOK);
  scatter_sum_kernel<<<dim3((NEXP * TPE) / 256), dim3(256), 0, stream>>>(gating, gidx, sums);
  norm_kernel<<<dim3((NEXP * TPE) / 256), dim3(256), 0, stream>>>(gating, gidx, sums);

  // shared expert: hs @ sgu -> x*silu(g) -> @ sdw -> dense store to out
  gemm1_swiglu<<<dim3(128 * (MDIM / 64), 1), dim3(256), 0, stream>>>(
      hs, nullptr, sgu, act, /*numRowBlocks=*/TOK / 128, /*siluFirst=*/0);
  gemm2_store<<<dim3((TOK / 128) * (HID / 128), 1), dim3(256), 0, stream>>>(
      act, sdw, out, TOK / 128);

  // routed experts: gather -> silu(g)*u -> down-proj -> scaled atomic combine
  gemm1_swiglu<<<dim3((TPE / 128) * (MDIM / 64), NEXP), dim3(256), 0, stream>>>(
      hs, gidx, gup, act, /*numRowBlocks=*/TPE / 128, /*siluFirst=*/1);
  gemm2_scatter<<<dim3((TPE / 128) * (HID / 128), NEXP), dim3(256), 0, stream>>>(
      act, dwn, gating, gidx, out, TPE / 128);
}
